// ScaledBinaryLinear_11218454577486
// MI455X (gfx1250) — matrix operand order repaired and checked by reading, not compiled, not run on hardware
//
#include <hip/hip_runtime.h>

// ---------------------------------------------------------------------------
// ScaledBinaryLinear: y = clip( (x @ sign(W)^T) * scale + bias, +-100 )
//   x: [4, 8192, 1024] f32  -> M = 32768, K = 1024
//   W: [1024, 1024]    f32  -> N = 1024 (binarized to +-1, exact in f16)
// Path: f32 -> f16 via v_cvt_pk_rtz_f16_f32 on LDS store,
//       sign(W) -> packed f16 via v_perm_b32 + v_and_or_b32 (2 VALU / pair),
//       V_WMMA_F32_16X16X32_F16, f32 accumulate.
// Block tile 128x128, 8 waves (4M x 2N), wave tile 32x64 = 8 WMMA / K-step.
// ---------------------------------------------------------------------------

typedef __attribute__((ext_vector_type(16))) _Float16 f16x16;
typedef __attribute__((ext_vector_type(2)))  __fp16   fp16x2_raw; // builtin ret type
typedef __attribute__((ext_vector_type(8)))  float    f32x8;

union FragU {
    f16x16 v;
    uint4  q[2];
};

// two f32 -> packed f16 pair, single VALU instruction (v_cvt_pk_rtz_f16_f32)
__device__ __forceinline__ unsigned pack_f16(float a, float b) {
    union { fp16x2_raw h; unsigned u; } cv;
    cv.h = __builtin_amdgcn_cvt_pkrtz(a, b);
    return cv.u;
}

// sign-STE binarization to packed f16 {+-1, +-1} in 2 VALU ops:
//   v_perm_b32: out.b1 = a.b3 (sign of a -> bit15), out.b3 = b.b3 (sign -> bit31)
//   then (w & 0x80008000) | 0x3C003C00  -> v_and_or_b32
__device__ __forceinline__ unsigned pack_sign_f16(float a, float b) {
    unsigned ua = __float_as_uint(a);
    unsigned ub = __float_as_uint(b);
    unsigned w  = __builtin_amdgcn_perm(ub, ua, 0x07000300u);
    return (w & 0x80008000u) | 0x3C003C00u;
}

#define M_TOT   32768
#define N_TOT   1024
#define K_TOT   1024
#define MT      128          // workgroup M tile
#define NTB     128          // workgroup N tile
#define KT      32           // K step (one f16 WMMA depth)
#define NKSTEP  (K_TOT / KT) // 32
#define LDS_STRIDE 40        // 32 + 8 halfword pad (16B-aligned, staggers banks)

__global__ __launch_bounds__(256)
void ScaledBinaryLinear_11218454577486_kernel(const float* __restrict__ x,
                                              const float* __restrict__ wgt,
                                              const float* __restrict__ scale,
                                              const float* __restrict__ bias,
                                              float* __restrict__ out) {
    __shared__ __align__(16) unsigned short ldsA[MT  * LDS_STRIDE]; // 10240 B
    __shared__ __align__(16) unsigned short ldsB[NTB * LDS_STRIDE]; // 10240 B

    const int tid = threadIdx.x;
    const int m0  = blockIdx.y * MT;
    const int n0  = blockIdx.x * NTB;

    // cooperative loader decomposition: each thread owns one float4 column slot
    const int lrow = tid >> 3;        // 0..31
    const int scol = (tid & 7) * 4;   // f32 column within K-step: 0,4,...,28

    // wave tiling: 8 waves as 4(M) x 2(N); each wave -> 32x64 (2x4 WMMA tiles)
    const int wave  = tid >> 5;
    const int lane  = tid & 31;
    const int waveM = wave & 3;       // 0..3 -> 32-row strips
    const int waveN = wave >> 2;      // 0..1 -> 64-col strips
    const int lr    = lane & 15;      // row/col within a 16x16 tile
    const int lh    = lane >> 4;      // K-half selector (CDNA5 16-bit layout)

    const float* pa = x   + (size_t)(m0 + lrow) * K_TOT + scol;
    const float* pb = wgt + (size_t)(n0 + lrow) * K_TOT + scol;

    // register prefetch buffers (double-buffer over single LDS buffer)
    float4 ra[4], rb[4];
#pragma unroll
    for (int i = 0; i < 4; ++i) ra[i] = *(const float4*)(pa + (size_t)(i * 32) * K_TOT);
#pragma unroll
    for (int i = 0; i < 4; ++i) rb[i] = *(const float4*)(pb + (size_t)(i * 32) * K_TOT);

    f32x8 acc[2][4];
#pragma unroll
    for (int mt = 0; mt < 2; ++mt)
#pragma unroll
        for (int nt = 0; nt < 4; ++nt)
            acc[mt][nt] = (f32x8){0.f, 0.f, 0.f, 0.f, 0.f, 0.f, 0.f, 0.f};

    for (int kk = 0; kk < NKSTEP; ++kk) {
        __syncthreads();   // previous step's LDS consumers are done

        // convert-on-store: f32 -> f16 (1 cvt_pk per pair) into LDS
#pragma unroll
        for (int i = 0; i < 4; ++i) {
            uint2 p;
            p.x = pack_f16(ra[i].x, ra[i].y);
            p.y = pack_f16(ra[i].z, ra[i].w);
            *(uint2*)&ldsA[(lrow + i * 32) * LDS_STRIDE + scol] = p;
        }
#pragma unroll
        for (int i = 0; i < 4; ++i) {
            uint2 p;
            p.x = pack_sign_f16(rb[i].x, rb[i].y);
            p.y = pack_sign_f16(rb[i].z, rb[i].w);
            *(uint2*)&ldsB[(lrow + i * 32) * LDS_STRIDE + scol] = p;
        }
        __syncthreads();

        // prefetch next K-step while WMMAs run (wave-uniform branch: EXEC full)
        if (kk + 1 < NKSTEP) {
            const int ko = (kk + 1) * KT;
#pragma unroll
            for (int i = 0; i < 4; ++i) ra[i] = *(const float4*)(pa + ko + (size_t)(i * 32) * K_TOT);
#pragma unroll
            for (int i = 0; i < 4; ++i) rb[i] = *(const float4*)(pb + ko + (size_t)(i * 32) * K_TOT);
        }

        // A fragments: lanes 0-15 hold K {0..7,16..23}, lanes 16-31 K {8..15,24..31}
        FragU fa[2], fb[4];
#pragma unroll
        for (int mt = 0; mt < 2; ++mt) {
            const int row = waveM * 32 + mt * 16 + lr;
            fa[mt].q[0] = *(const uint4*)&ldsA[row * LDS_STRIDE + lh * 8];
            fa[mt].q[1] = *(const uint4*)&ldsA[row * LDS_STRIDE + 16 + lh * 8];
        }
        // B fragments: lane holds column n; lanes 0-15 K {0..7,16..23}, lanes 16-31 K {8..15,24..31}
#pragma unroll
        for (int nt = 0; nt < 4; ++nt) {
            const int col = waveN * 64 + nt * 16 + lr;
            fb[nt].q[0] = *(const uint4*)&ldsB[col * LDS_STRIDE + lh * 8];
            fb[nt].q[1] = *(const uint4*)&ldsB[col * LDS_STRIDE + 16 + lh * 8];
        }

#pragma unroll
        for (int mt = 0; mt < 2; ++mt)
#pragma unroll
            for (int nt = 0; nt < 4; ++nt)
                acc[mt][nt] = __builtin_amdgcn_wmma_f32_16x16x32_f16(
                    /*neg_a=*/false, fa[mt].v,
                    /*neg_b=*/false, fb[nt].v,
                    /*c_mod=*/(short)0, acc[mt][nt],
                    /*reuse_a=*/false, /*reuse_b=*/false);
    }

    // Epilogue: y = clip(acc*scale + bias). C layout: N = lane%16 (per-lane const),
    // M = vgpr + 8*(lane/16) within the 16x16 tile.
#pragma unroll
    for (int nt = 0; nt < 4; ++nt) {
        const int col = n0 + waveN * 64 + nt * 16 + lr;
        const float s  = scale[col];
        const float bv = bias[col];
#pragma unroll
        for (int mt = 0; mt < 2; ++mt) {
            const int rbase = m0 + waveM * 32 + mt * 16 + lh * 8;
#pragma unroll
            for (int v = 0; v < 8; ++v) {
                float y = acc[mt][nt][v] * s + bv;
                y = fminf(fmaxf(y, -100.0f), 100.0f);
                out[(size_t)(rbase + v) * N_TOT + col] = y;
            }
        }
    }
}

extern "C" void kernel_launch(void* const* d_in, const int* in_sizes, int n_in,
                              void* d_out, int out_size, void* d_ws, size_t ws_size,
                              hipStream_t stream) {
    const float* x     = (const float*)d_in[0];   // [4, 8192, 1024]
    const float* wgt   = (const float*)d_in[1];   // [1024, 1024]
    const float* scale = (const float*)d_in[2];   // [1024]
    const float* bias  = (const float*)d_in[3];   // [1024]
    float* out = (float*)d_out;                   // [4, 8192, 1024]

    dim3 grid(N_TOT / NTB, M_TOT / MT, 1);        // (8, 256)
    dim3 block(256, 1, 1);                        // 8 waves (wave32)
    ScaledBinaryLinear_11218454577486_kernel<<<grid, block, 0, stream>>>(
        x, wgt, scale, bias, out);
}
